// SpanPairVAHead_67860483277731
// MI455X (gfx1250) — matrix operand order repaired and checked by reading, not compiled, not run on hardware
//
#include <hip/hip_runtime.h>
#include <hip/hip_bf16.h>
#include <math.h>

typedef __attribute__((ext_vector_type(8)))  _Float16 v8h;
typedef __attribute__((ext_vector_type(16))) _Float16 v16h;
typedef __attribute__((ext_vector_type(8)))  float    v8f;

#define NB   32      // batch
#define NL   512     // sequence length
#define NH   1024    // hidden
#define NQ   128     // queries
#define NM   (NB*NQ) // 4096 rows
#define K1   (3*NH)  // 3072
#define N1   256     // VA_H
#define N2   128     // VA_H/2

// ---------------------------------------------------------------------------
// Phase 1: span mean-pooling.  One block per (b,q).  Reads <=16 rows of H=1024
// f32 each; writes the 3H-wide f16 feature row [h_asp | h_opi | h_asp*h_opi].
// This phase carries the compulsory 64MB HBM traffic (span re-reads hit L2).
// ---------------------------------------------------------------------------
__global__ void pool_kernel(const float* __restrict__ hs,
                            const int*  __restrict__ spans,
                            _Float16*   __restrict__ Xp) {
  int bq = blockIdx.x;                 // 0..NM-1  (= b*NQ + q)
  int b  = bq >> 7;                    // / NQ
  const int* sp = spans + (size_t)bq * 4;
  int as = sp[0], ae = sp[1], os = sp[2], oe = sp[3];
  const float* base = hs + (size_t)b * NL * NH;
  bool aok = (as >= 2) && (ae >= as);
  bool ook = (os >= 2) && (oe >= os);
  float ia = aok ? 1.0f / (float)(ae - as + 1) : 0.0f;
  float io = ook ? 1.0f / (float)(oe - os + 1) : 0.0f;
  size_t row = (size_t)bq * (3 * NH);
  for (int h = threadIdx.x; h < NH; h += blockDim.x) {
    float a;
    if (aok) { float s = 0.f; for (int p = as; p <= ae; ++p) s += base[(size_t)p * NH + h]; a = s * ia; }
    else     { a = base[NH + h]; }     // sep token = row 1
    float o;
    if (ook) { float s = 0.f; for (int p = os; p <= oe; ++p) s += base[(size_t)p * NH + h]; o = s * io; }
    else     { o = base[NH + h]; }
    Xp[row + h]          = (_Float16)a;
    Xp[row + NH + h]     = (_Float16)o;
    Xp[row + 2 * NH + h] = (_Float16)(a * o);
  }
}

// ---------------------------------------------------------------------------
// f32 -> f16 weight conversion (W1, W2 become WMMA B-operands)
// ---------------------------------------------------------------------------
__global__ void f32_to_f16_kernel(const float* __restrict__ in,
                                  _Float16* __restrict__ out, int n) {
  int i = blockIdx.x * blockDim.x + threadIdx.x;
  if (i < n) out[i] = (_Float16)in[i];
}

__device__ __forceinline__ float gelu_exact(float x) {
  return 0.5f * x * (1.0f + erff(x * 0.70710678118654752f));
}

// Fragment loaders (wave32); this code's own contiguous k order, same for A and B:
//   A lane(m=l&15, kh=l>>4): elems 0..15 = K kh*16+0..15 (contiguous)
//   B lane(n=l&15, kh):      elems 0..15 = K kh*16+0..15 (contiguous)
__device__ __forceinline__ v16h load_a_frag(const _Float16* Arow, int kk, int kh) {
  v8h lo = *(const v8h*)(Arow + kk + kh * 16);
  v8h hi = *(const v8h*)(Arow + kk + kh * 16 + 8);
  return __builtin_shufflevector(lo, hi,
                                 0, 1, 2, 3, 4, 5, 6, 7,
                                 8, 9, 10, 11, 12, 13, 14, 15);
}
__device__ __forceinline__ v16h load_b_frag(const _Float16* Wrow, int kk, int kh) {
  return *(const v16h*)(Wrow + kk + kh * 16);
}

#define WMMA_F16(ACC, AF, BF)                                              \
  ACC = __builtin_amdgcn_wmma_f32_16x16x32_f16(false, (AF), false, (BF),   \
                                               (short)0, (ACC), false, false)

// ---------------------------------------------------------------------------
// Out = gelu(A @ W^T + bias) via v_wmma_f32_16x16x32_f16.
// Register double-buffered with a PEELED final step: the steady-state loop
// always loads kk+32 (in-bounds), so load addresses are affine in kk and the
// compiler can use one induction base + immediate offsets in s_clause groups,
// while next-step loads stay in flight across the current 4-WMMA group.
// A: MxK row-major f16.  W: NxK row-major f16 (row n == column n of W^T).
// grid = (M/64, N/64), block = (32,4): wave ty owns M-tile bx*4+ty, 16x64 strip.
// C layout: lane l, vgpr r -> n = l&15, m = (l>>4)*8 + r.
// ---------------------------------------------------------------------------
__global__ __launch_bounds__(128) void gemm_gelu_wmma(
    const _Float16* __restrict__ A, const _Float16* __restrict__ W,
    const float* __restrict__ bias, _Float16* __restrict__ Out,
    int M, int N, int K) {
  int lane = threadIdx.x;
  int l15  = lane & 15;
  int kh   = lane >> 4;
  int mt   = blockIdx.x * 4 + threadIdx.y;   // M tile index
  int nb   = blockIdx.y * 64;                // N base column
  const _Float16* Arow = A + (size_t)(mt * 16 + l15) * K;
  const _Float16* Wr0 = W + (size_t)(nb + 0 * 16 + l15) * K;
  const _Float16* Wr1 = W + (size_t)(nb + 1 * 16 + l15) * K;
  const _Float16* Wr2 = W + (size_t)(nb + 2 * 16 + l15) * K;
  const _Float16* Wr3 = W + (size_t)(nb + 3 * 16 + l15) * K;

  v8f acc0 = {}, acc1 = {}, acc2 = {}, acc3 = {};

  // Prologue: fragments for kk = 0
  v16h a  = load_a_frag(Arow, 0, kh);
  v16h b0 = load_b_frag(Wr0, 0, kh);
  v16h b1 = load_b_frag(Wr1, 0, kh);
  v16h b2 = load_b_frag(Wr2, 0, kh);
  v16h b3 = load_b_frag(Wr3, 0, kh);

  // Steady state: issue next-step loads (affine addresses, no wrap), then
  // the current 4-WMMA group; rotate register buffers.
  for (int kk = 0; kk < K - 32; kk += 32) {
    v16h an  = load_a_frag(Arow, kk + 32, kh);
    v16h b0n = load_b_frag(Wr0, kk + 32, kh);
    v16h b1n = load_b_frag(Wr1, kk + 32, kh);
    v16h b2n = load_b_frag(Wr2, kk + 32, kh);
    v16h b3n = load_b_frag(Wr3, kk + 32, kh);
    __builtin_prefetch(Arow + kk + 512, 0, 1);  // global_prefetch_b8 ahead

    WMMA_F16(acc0, a, b0);
    WMMA_F16(acc1, a, b1);
    WMMA_F16(acc2, a, b2);
    WMMA_F16(acc3, a, b3);

    a = an; b0 = b0n; b1 = b1n; b2 = b2n; b3 = b3n;
  }

  // Peeled final K-step: no trailing loads
  WMMA_F16(acc0, a, b0);
  WMMA_F16(acc1, a, b1);
  WMMA_F16(acc2, a, b2);
  WMMA_F16(acc3, a, b3);

  // Epilogue: bias + exact GELU, scatter f16 stores per the C layout
  int mbase = mt * 16 + kh * 8;
#pragma unroll
  for (int r = 0; r < 8; ++r) {
    int mrow = mbase + r;
    int c0 = nb + 0 * 16 + l15;
    int c1 = nb + 1 * 16 + l15;
    int c2 = nb + 2 * 16 + l15;
    int c3 = nb + 3 * 16 + l15;
    Out[(size_t)mrow * N + c0] = (_Float16)gelu_exact(acc0[r] + bias[c0]);
    Out[(size_t)mrow * N + c1] = (_Float16)gelu_exact(acc1[r] + bias[c1]);
    Out[(size_t)mrow * N + c2] = (_Float16)gelu_exact(acc2[r] + bias[c2]);
    Out[(size_t)mrow * N + c3] = (_Float16)gelu_exact(acc3[r] + bias[c3]);
  }
}

// ---------------------------------------------------------------------------
// Phase 4: tiny head  va = sigmoid(x2 @ W3^T + b3)*8+1, masked.  N=2 -> scalar.
// ---------------------------------------------------------------------------
__global__ void head_kernel(const _Float16* __restrict__ X2,
                            const float* __restrict__ W3,
                            const float* __restrict__ b3,
                            const float* __restrict__ mask,
                            float* __restrict__ out) {
  int m = blockIdx.x * blockDim.x + threadIdx.x;
  if (m >= NM) return;
  const _Float16* xr = X2 + (size_t)m * N2;
  float msk = (mask[m] >= 0.5f) ? 1.0f : 0.0f;
#pragma unroll
  for (int n = 0; n < 2; ++n) {
    const float* wr = W3 + n * N2;
    float s = b3[n];
    for (int k = 0; k < N2; ++k) s += (float)xr[k] * wr[k];
    float sg = 1.0f / (1.0f + expf(-s));
    out[(size_t)m * 2 + n] = (sg * 8.0f + 1.0f) * msk;
  }
}

extern "C" void kernel_launch(void* const* d_in, const int* in_sizes, int n_in,
                              void* d_out, int out_size, void* d_ws, size_t ws_size,
                              hipStream_t stream) {
  const float* hs    = (const float*)d_in[0];   // (B,L,H) f32
  const int*   spans = (const int*)  d_in[1];   // (B,Q,4) int
  const float* qmask = (const float*)d_in[2];   // (B,Q)   f32
  const float* W1    = (const float*)d_in[3];   // (256,3072)
  const float* b1    = (const float*)d_in[4];   // (256,)
  const float* W2    = (const float*)d_in[5];   // (128,256)
  const float* b2    = (const float*)d_in[6];   // (128,)
  const float* W3    = (const float*)d_in[7];   // (2,128)
  const float* b3    = (const float*)d_in[8];   // (2,)
  float* out = (float*)d_out;                   // (B,Q,2) f32

  // Workspace layout (f16 halves), ~29MB total
  _Float16* ws16  = (_Float16*)d_ws;
  _Float16* Xpair = ws16;                               // NM * K1
  _Float16* W1h   = Xpair + (size_t)NM * K1;            // N1 * K1
  _Float16* X1    = W1h   + (size_t)N1 * K1;            // NM * N1
  _Float16* W2h   = X1    + (size_t)NM * N1;            // N2 * N1
  _Float16* X2    = W2h   + (size_t)N2 * N1;            // NM * N2

  // 1) pooling (HBM-bound: streams 64MB of hidden_states once)
  pool_kernel<<<NM, 256, 0, stream>>>(hs, spans, Xpair);

  // 2) weight conversion (L2-resident, ~3MB)
  f32_to_f16_kernel<<<(N1 * K1 + 255) / 256, 256, 0, stream>>>(W1, W1h, N1 * K1);
  f32_to_f16_kernel<<<(N2 * N1 + 255) / 256, 256, 0, stream>>>(W2, W2h, N2 * N1);

  // 3) GEMM1: (4096 x 3072) @ (3072 x 256) + gelu   -> grid (64, 4)
  gemm_gelu_wmma<<<dim3(NM / 64, N1 / 64), dim3(32, 4), 0, stream>>>(
      Xpair, W1h, b1, X1, NM, N1, K1);

  // 4) GEMM2: (4096 x 256) @ (256 x 128) + gelu     -> grid (64, 2)
  gemm_gelu_wmma<<<dim3(NM / 64, N2 / 64), dim3(32, 4), 0, stream>>>(
      X1, W2h, b2, X2, NM, N2, N1);

  // 5) head: sigmoid*8+1, mask
  head_kernel<<<(NM + 255) / 256, 256, 0, stream>>>(X2, W3, b3, qmask, out);
}
